// LocalDPT_51702816309479
// MI455X (gfx1250) — matrix operand order repaired and checked by reading, not compiled, not run on hardware
//
#include <hip/hip_runtime.h>
#include <hip/hip_bf16.h>

// ---------------------------------------------------------------------------
// LocalDPT forward for gfx1250 (MI455X).
// Implicit-GEMM convs via v_wmma_f32_16x16x32_f16; 1 workgroup (8 wave32) per
// query; f16 activations in LDS; weights pre-packed to B-fragment layout in
// d_ws by a prep kernel. Each wave computes two N-tiles per A fragment with
// software-pipelined B loads.
// ---------------------------------------------------------------------------

typedef __attribute__((ext_vector_type(16))) _Float16 v16h;
typedef __attribute__((ext_vector_type(8)))  _Float16 v8h;
typedef __attribute__((ext_vector_type(8)))  short    v8s;
typedef __attribute__((ext_vector_type(8)))  float    v8f;

#define D_CH 64
#define BQ   2
#define KQ   2048
#define NTHREADS 256
#define NWAVES 8

// packed-weight offsets (in halves) inside d_ws
#define SZ_RCUW (18 * 4 * 512)   // ktiles=18, ntiles=4, 32 lanes * 16 halves
#define SZ_RNW  (2 * 4 * 512)
#define SZ_OC1W (18 * 2 * 512)
#define OFF_S8W1 0
#define OFF_S8W2 (OFF_S8W1 + SZ_RCUW)
#define OFF_L2W1 (OFF_S8W2 + SZ_RCUW)
#define OFF_L2W2 (OFF_L2W1 + SZ_RCUW)
#define OFF_S4W1 (OFF_L2W2 + SZ_RCUW)
#define OFF_S4W2 (OFF_S4W1 + SZ_RCUW)
#define OFF_L1W1 (OFF_S4W2 + SZ_RCUW)
#define OFF_L1W2 (OFF_L1W1 + SZ_RCUW)
#define OFF_RN2  (OFF_L1W2 + SZ_RCUW)
#define OFF_RN1  (OFF_RN2 + SZ_RNW)
#define OFF_OC1  (OFF_RN1 + SZ_RNW)

// ---------------------------------------------------------------------------
// Weight prep: OIHW f32 -> f16, laid out per B-fragment:
// dst[((kt*ntiles + nt)*32 + lane)*16 + e]
// B 32x16 layout: col = lane&15; h = lane>>4; e<8 -> K=8h+e,
// e>=8 -> K=16+8h+(e-8).  K index = tap*64 + c.
// ---------------------------------------------------------------------------
__global__ void ldpt_pack_weights(const float* __restrict__ W,
                                  _Float16* __restrict__ dst,
                                  int O, int taps, int ktiles, int ntiles) {
    int idx = blockIdx.x * blockDim.x + threadIdx.x;
    int total = ktiles * ntiles * 512;
    if (idx >= total) return;
    int e    = idx & 15;
    int lane = (idx >> 4) & 31;
    int t    = idx >> 9;            // kt*ntiles + nt
    int kt   = t / ntiles;
    int nt   = t - kt * ntiles;
    int o    = nt * 16 + (lane & 15);
    int K    = kt * 32 + ((e >> 3) << 4) + ((lane >> 4) << 3) + (e & 7);
    int KT   = taps * D_CH;
    _Float16 val = (_Float16)0.f;
    if (o < O && K < KT) {
        int tap = K >> 6;           // K / 64
        int c   = K & 63;
        val = (_Float16)W[(o * D_CH + c) * taps + tap];
    }
    dst[idx] = val;
}

// ---------------------------------------------------------------------------
// helpers
// ---------------------------------------------------------------------------
__device__ __forceinline__ void zero_lds(_Float16* p, int n) {
    for (int i = threadIdx.x; i < n; i += NTHREADS) p[i] = (_Float16)0.f;
}

// packed relu on 8 halves: bits & ~(bits >> 15)  (2 packed int ops)
__device__ __forceinline__ v8h relu8(v8h x) {
    v8s b = __builtin_bit_cast(v8s, x);
    b = b & ~(b >> 15);
    return __builtin_bit_cast(v8h, b);
}

__device__ __forceinline__ v16h pack16(v8h lo, v8h hi) {
    v16h r;
#pragma unroll
    for (int i = 0; i < 8; ++i) { r[i] = lo[i]; r[i + 8] = hi[i]; }
    return r;
}

// Bilinear-sample an N x N patch (64 channels) centered at (cx, cy), zeros
// padding, into LDS at dst[(y*DPW + x)*64 + c]. Channel-outer iteration so
// adjacent lanes touch adjacent pixels (better vmem coalescing).
template <int N, int HW, int DPW>
__device__ __forceinline__ void sample_patch(const float* __restrict__ img,
                                             float cx, float cy,
                                             _Float16* dst) {
    constexpr int NN   = N * N;
    constexpr int half = N / 2;
    for (int idx = threadIdx.x; idx < D_CH * NN; idx += NTHREADS) {
        int c  = idx / NN;
        int sp = idx - c * NN;
        int y  = sp / N;
        int x  = sp - y * N;
        float sx = cx + (float)(x - half);
        float sy = cy + (float)(y - half);
        float x0f = floorf(sx), y0f = floorf(sy);
        float wx = sx - x0f, wy = sy - y0f;
        int x0 = (int)x0f, y0 = (int)y0f;
        float v = 0.f;
#pragma unroll
        for (int dy = 0; dy < 2; ++dy)
#pragma unroll
            for (int dx = 0; dx < 2; ++dx) {
                int xi = x0 + dx, yi = y0 + dy;
                float w = (dx ? wx : 1.f - wx) * (dy ? wy : 1.f - wy);
                if (xi >= 0 && xi < HW && yi >= 0 && yi < HW)
                    v += w * img[(c * HW + yi) * HW + xi];
            }
        dst[(y * DPW + x) * D_CH + c] = (_Float16)v;
    }
}

// Fused bilinear 2x upsample (align_corners=True) + recenter-crop.
// src: SSRC x SSRC x C (stride SSRC) -> dst: CROP x CROP x C (stride CROP).
template <int SSRC, int C, int CROP>
__device__ __forceinline__ void resize_crop(const _Float16* __restrict__ src,
                                            _Float16* dst, int ox, int oy) {
    constexpr int R  = 2 * SSRC;
    constexpr int cb = R / 2 - 1 - CROP / 2;
    constexpr float scale = (float)(SSRC - 1) / (float)(R - 1);
    for (int idx = threadIdx.x; idx < CROP * CROP * C; idx += NTHREADS) {
        int c  = idx % C;
        int sp = idx / C;
        int x  = sp % CROP;
        int y  = sp / CROP;
        float sy = (float)(cb + oy + y) * scale;
        float sx = (float)(cb + ox + x) * scale;
        int y0 = (int)sy, x0 = (int)sx;
        float wy = sy - (float)y0, wx = sx - (float)x0;
        int y1 = min(y0 + 1, SSRC - 1), x1 = min(x0 + 1, SSRC - 1);
        float v00 = (float)src[(y0 * SSRC + x0) * C + c];
        float v01 = (float)src[(y0 * SSRC + x1) * C + c];
        float v10 = (float)src[(y1 * SSRC + x0) * C + c];
        float v11 = (float)src[(y1 * SSRC + x1) * C + c];
        float v = (1.f - wy) * ((1.f - wx) * v00 + wx * v01) +
                  wy * ((1.f - wx) * v10 + wx * v11);
        dst[(y * CROP + x) * C + c] = (_Float16)v;
    }
}

// ---------------------------------------------------------------------------
// Implicit-GEMM conv via WMMA f32_16x16x32_f16, fully specialized shapes.
// act:  LDS base so output (y,x) reads act[((y+dy)*APW + (x+dx))*64 + c].
// Bp:   packed B fragments (global).  dest[(y*DPW + x)*DC + n].
// Each wave owns (mtile, ntile-pair): one A fragment feeds two WMMAs.
// ---------------------------------------------------------------------------
template <int SOUT, int APW, int TAPS, bool RELU, int COUT, int DPW, int DC,
          bool ACCUM, bool HASRES, int RPW>
__device__ __forceinline__ void conv_wmma(const _Float16* __restrict__ act,
                                          const _Float16* __restrict__ Bp,
                                          const float* __restrict__ bias,
                                          const _Float16* __restrict__ resid,
                                          _Float16* __restrict__ dest) {
    constexpr int M      = SOUT * SOUT;
    constexpr int nM     = (M + 15) / 16;
    constexpr int nN     = COUT / 16;      // 4 or 2 (always even)
    constexpr int nPair  = nN / 2;
    constexpr int ksteps = (TAPS == 9) ? 18 : 2;
    constexpr int BSTR   = nN * 512;       // halves between k-steps

    const int lane = threadIdx.x & 31;
    const int wave = threadIdx.x >> 5;
    const int mrow = lane & 15;
    const int kb8  = (lane >> 4) << 3;     // 0 or 8 (A K-sublane base)

    for (int t = wave; t < nM * nPair; t += NWAVES) {
        int mt  = t / nPair;
        int np  = t - mt * nPair;
        int nt0 = np * 2, nt1 = nt0 + 1;
        v8f acc0 = {}, acc1 = {};
        int m  = mt * 16 + mrow;
        int mc = (m < M) ? m : (M - 1);    // clamp: garbage rows never stored
        int my = mc / SOUT, mx = mc - my * SOUT;
        const _Float16* abase = &act[(my * APW + mx) * D_CH + kb8];

        const _Float16* bp0 = Bp + (nt0 * 32 + lane) * 16;
        const _Float16* bp1 = Bp + (nt1 * 32 + lane) * 16;
        v8h b0l = *(const v8h*)bp0, b0h = *(const v8h*)(bp0 + 8);
        v8h b1l = *(const v8h*)bp1, b1h = *(const v8h*)(bp1 + 8);

        for (int ks = 0; ks < ksteps; ++ks) {
            int cbase = (ks & 1) << 5;     // 0 or 32
            int off;
            if constexpr (TAPS == 9) {
                int tap = ks >> 1;
                int dy  = tap / 3;
                int dx  = tap - 3 * dy;
                off = (dy * APW + dx) * D_CH + cbase;
            } else {
                off = cbase;
            }
            const _Float16* p = abase + off;
            v8h c0 = *(const v8h*)p;        // channels +0..7
            v8h c1 = *(const v8h*)(p + 16); // channels +16..23
            if constexpr (RELU) { c0 = relu8(c0); c1 = relu8(c1); }
            v16h a   = pack16(c0, c1);
            v16h bm0 = pack16(b0l, b0h);
            v16h bm1 = pack16(b1l, b1h);
            if (ks + 1 < ksteps) {          // pipeline next B fragments
                bp0 += BSTR; bp1 += BSTR;
                b0l = *(const v8h*)bp0; b0h = *(const v8h*)(bp0 + 8);
                b1l = *(const v8h*)bp1; b1h = *(const v8h*)(bp1 + 8);
            }
            acc0 = __builtin_amdgcn_wmma_f32_16x16x32_f16(
                false, a, false, bm0, (short)0, acc0, false, false);
            acc1 = __builtin_amdgcn_wmma_f32_16x16x32_f16(
                false, a, false, bm1, (short)0, acc1, false, false);
        }
        // epilogue: lane l -> col = l&15; vgpr i -> row = (l>>4)*8 + i
        int mbase = mt * 16 + ((lane >> 4) << 3);
#pragma unroll
        for (int half = 0; half < 2; ++half) {
            int   nt  = half ? nt1 : nt0;
            v8f&  acc = half ? acc1 : acc0;
            int   n   = nt * 16 + (lane & 15);
            float bv  = bias[n];
#pragma unroll
            for (int i = 0; i < 8; ++i) {
                int mm = mbase + i;
                if (mm < M) {
                    int yy = mm / SOUT, xx = mm - yy * SOUT;
                    float val = acc[i] + bv;
                    if constexpr (HASRES)
                        val += (float)resid[(yy * RPW + xx) * D_CH + n];
                    _Float16* dp = &dest[(yy * DPW + xx) * DC + n];
                    if constexpr (ACCUM) val += (float)(*dp);
                    *dp = (_Float16)val;
                }
            }
        }
    }
}

// ---------------------------------------------------------------------------
// Main kernel: one workgroup per query.
// ---------------------------------------------------------------------------
#define PBUF (11 * 11 * D_CH)

__global__ __launch_bounds__(NTHREADS) void ldpt_forward(
    const float* __restrict__ s8, const float* __restrict__ L1,
    const float* __restrict__ L2, const float* __restrict__ coords,
    const _Float16* __restrict__ wp,
    const float* __restrict__ b_s8_1, const float* __restrict__ b_s8_2,
    const float* __restrict__ b_l2_1, const float* __restrict__ b_l2_2,
    const float* __restrict__ b_s4_1, const float* __restrict__ b_s4_2,
    const float* __restrict__ b_l1_1, const float* __restrict__ b_l1_2,
    const float* __restrict__ b_rn2, const float* __restrict__ b_rn1,
    const float* __restrict__ b_oc1,
    const float* __restrict__ oc2a_w, const float* __restrict__ oc2a_b,
    const float* __restrict__ oc2c_w, const float* __restrict__ oc2c_b,
    float* __restrict__ out) {
    __shared__ _Float16 bufA[PBUF];
    __shared__ _Float16 bufB[PBUF];
    __shared__ _Float16 bufC[PBUF];
    __shared__ _Float16 bufS[9 * 9 * D_CH];
    __shared__ float red[32];

    const int q = blockIdx.x;
    const int b = q >> 11;     // / 2048
    const int k = q & 2047;
    const float u = coords[(b * KQ + k) * 2 + 0];
    const float v = coords[(b * KQ + k) * 2 + 1];
    const int ox8 = (fmodf(u, 8.f) >= 4.f) ? 1 : 0;
    const int oy8 = (fmodf(v, 8.f) >= 4.f) ? 1 : 0;
    const int ox4 = (fmodf(u, 4.f) >= 2.f) ? 1 : 0;
    const int oy4 = (fmodf(v, 4.f) >= 2.f) ? 1 : 0;
    const int ox2 = (fmodf(u, 2.f) >= 1.f) ? 1 : 0;
    const int oy2 = (fmodf(v, 2.f) >= 1.f) ? 1 : 0;

    const float* s8b = s8 + (size_t)b * D_CH * 128 * 128;
    const float* L2b = L2 + (size_t)b * D_CH * 128 * 128;
    const float* L1b = L1 + (size_t)b * D_CH * 256 * 256;
    const int O11 = (11 + 1) * D_CH;   // offset of logical (1,1) in padded buf

    // ---------------- stage 1: p8 RCU ----------------
    zero_lds(bufA, PBUF);
    zero_lds(bufB, PBUF);
    __syncthreads();
    sample_patch<9, 128, 11>(s8b, u * 0.125f, v * 0.125f, bufA + O11);
    __syncthreads();
    conv_wmma<9, 11, 9, true, 64, 11, 64, false, false, 0>(
        bufA, wp + OFF_S8W1, b_s8_1, nullptr, bufB + O11);
    __syncthreads();
    conv_wmma<9, 11, 9, true, 64, 9, 64, false, true, 11>(
        bufB, wp + OFF_S8W2, b_s8_2, bufA + O11, bufS);
    __syncthreads();
    // ---------------- stage 1: pl2 RCU, accumulate into bufS -------------
    zero_lds(bufA, PBUF);
    __syncthreads();
    sample_patch<9, 128, 11>(L2b, u * 0.125f, v * 0.125f, bufA + O11);
    __syncthreads();
    conv_wmma<9, 11, 9, true, 64, 11, 64, false, false, 0>(
        bufA, wp + OFF_L2W1, b_l2_1, nullptr, bufB + O11);
    __syncthreads();
    conv_wmma<9, 11, 9, true, 64, 9, 64, true, true, 11>(
        bufB, wp + OFF_L2W2, b_l2_2, bufA + O11, bufS);
    __syncthreads();
    // resize 9->18, crop 7 ; rn2 (1x1) -> padded bufB
    resize_crop<9, 64, 7>(bufS, bufC, ox8, oy8);
    __syncthreads();
    zero_lds(bufB, PBUF);
    __syncthreads();
    conv_wmma<7, 7, 1, false, 64, 11, 64, false, false, 0>(
        bufC, wp + OFF_RN2, b_rn2, nullptr, bufB + O11);
    __syncthreads();

    // ---------------- stage 2: s4 RCU + pl1 RCU ----------------
    zero_lds(bufA, PBUF);
    zero_lds(bufC, PBUF);
    __syncthreads();
    sample_patch<7, 256, 11>(L1b, u * 0.25f, v * 0.25f, bufA + O11);
    __syncthreads();
    conv_wmma<7, 11, 9, true, 64, 11, 64, false, false, 0>(
        bufB, wp + OFF_S4W1, b_s4_1, nullptr, bufC + O11);
    __syncthreads();
    conv_wmma<7, 11, 9, true, 64, 7, 64, false, true, 11>(
        bufC, wp + OFF_S4W2, b_s4_2, bufB + O11, bufS);
    __syncthreads();
    conv_wmma<7, 11, 9, true, 64, 11, 64, false, false, 0>(
        bufA, wp + OFF_L1W1, b_l1_1, nullptr, bufC + O11);
    __syncthreads();
    conv_wmma<7, 11, 9, true, 64, 7, 64, true, true, 11>(
        bufC, wp + OFF_L1W2, b_l1_2, bufA + O11, bufS);
    __syncthreads();
    // resize 7->14, crop 5 -> bufA (5x5x64, stride 5)
    resize_crop<7, 64, 5>(bufS, bufA, ox4, oy4);
    __syncthreads();
    // rn1 (1x1) -> bufB (5x5x64, stride 5)
    conv_wmma<5, 5, 1, false, 64, 5, 64, false, false, 0>(
        bufA, wp + OFF_RN1, b_rn1, nullptr, bufB);
    __syncthreads();
    // oc1 (3x3, valid): 5x5x64 -> 3x3x32 in bufC
    conv_wmma<3, 5, 9, false, 32, 3, 32, false, false, 0>(
        bufB, wp + OFF_OC1, b_oc1, nullptr, bufC);
    __syncthreads();
    // resize 3->6, crop 3 -> bufS (3x3x32)
    resize_crop<3, 32, 3>(bufC, bufS, ox2, oy2);
    __syncthreads();

    // ---------------- tail: oc2a (3x3 valid -> 1x1, relu) + oc2c + exp ----
    if (threadIdx.x < 32) {
        int o = threadIdx.x;
        float acc = oc2a_b[o];
#pragma unroll
        for (int tap = 0; tap < 9; ++tap) {
            int dy = tap / 3, dx = tap - 3 * dy;
            for (int c = 0; c < 32; ++c)
                acc += (float)bufS[tap * 32 + c] *
                       oc2a_w[((o * 32 + c) * 3 + dy) * 3 + dx];
        }
        acc = fmaxf(acc, 0.f);
        red[o] = acc * oc2c_w[o];
    }
    __syncthreads();
    if (threadIdx.x == 0) {
        float s = oc2c_b[0];
        for (int i = 0; i < 32; ++i) s += red[i];
        out[q] = expf(s);
    }
}

// ---------------------------------------------------------------------------
extern "C" void kernel_launch(void* const* d_in, const int* in_sizes, int n_in,
                              void* d_out, int out_size, void* d_ws,
                              size_t ws_size, hipStream_t stream) {
    const float* s8     = (const float*)d_in[0];
    const float* L1     = (const float*)d_in[1];
    const float* L2     = (const float*)d_in[2];
    const float* coords = (const float*)d_in[3];
    _Float16* wp = (_Float16*)d_ws;

    auto pack = [&](int in_idx, int off, int O, int taps, int kt, int nt) {
        int total = kt * nt * 512;
        ldpt_pack_weights<<<(total + 255) / 256, 256, 0, stream>>>(
            (const float*)d_in[in_idx], wp + off, O, taps, kt, nt);
    };
    pack(4,  OFF_S8W1, 64, 9, 18, 4);
    pack(6,  OFF_S8W2, 64, 9, 18, 4);
    pack(8,  OFF_L2W1, 64, 9, 18, 4);
    pack(10, OFF_L2W2, 64, 9, 18, 4);
    pack(12, OFF_S4W1, 64, 9, 18, 4);
    pack(14, OFF_S4W2, 64, 9, 18, 4);
    pack(16, OFF_L1W1, 64, 9, 18, 4);
    pack(18, OFF_L1W2, 64, 9, 18, 4);
    pack(20, OFF_RN2,  64, 1, 2, 4);
    pack(22, OFF_RN1,  64, 1, 2, 4);
    pack(24, OFF_OC1,  32, 9, 18, 2);

    ldpt_forward<<<BQ * KQ, NTHREADS, 0, stream>>>(
        s8, L1, L2, coords, wp,
        (const float*)d_in[5],  (const float*)d_in[7],
        (const float*)d_in[9],  (const float*)d_in[11],
        (const float*)d_in[13], (const float*)d_in[15],
        (const float*)d_in[17], (const float*)d_in[19],
        (const float*)d_in[21], (const float*)d_in[23],
        (const float*)d_in[25],
        (const float*)d_in[26], (const float*)d_in[27],
        (const float*)d_in[28], (const float*)d_in[29],
        (float*)d_out);
}
